// MuGuidedAttention_15092515078648
// MI455X (gfx1250) — matrix operand order repaired and checked by reading, not compiled, not run on hardware
//
#include <hip/hip_runtime.h>
#include <hip/hip_bf16.h>

// ---------------------------------------------------------------------------
// CDNA5 (gfx1250) WMMA types & helpers
// ---------------------------------------------------------------------------
typedef __attribute__((ext_vector_type(16))) __bf16          v16bf;
typedef __attribute__((ext_vector_type(8)))  float           v8f;
typedef __attribute__((ext_vector_type(8)))  unsigned short  ushort8;

union Frag16 {
    v16bf          v;
    ushort8        u8[2];
    unsigned short s[16];
    unsigned int   u32[8];
};

// float -> bf16 bits, round-half-up (1 add + 1 shift)
__device__ __forceinline__ unsigned short f2bf(float f) {
    unsigned int u = __builtin_bit_cast(unsigned int, f) + 0x8000u;
    return (unsigned short)(u >> 16);
}

// packed pair {bf16(lo), bf16(hi)} in one dword: 2x v_add + 1x v_perm_b32
__device__ __forceinline__ unsigned int f2bf2(float lo, float hi) {
    unsigned int ulo = __builtin_bit_cast(unsigned int, lo) + 0x8000u;
    unsigned int uhi = __builtin_bit_cast(unsigned int, hi) + 0x8000u;
#if __has_builtin(__builtin_amdgcn_perm)
    // out bytes {3,2} = uhi bytes {3,2}; out bytes {1,0} = ulo bytes {3,2}
    return __builtin_amdgcn_perm(uhi, ulo, 0x07060302u);
#else
    return (ulo >> 16) | (uhi & 0xFFFF0000u);
#endif
}

__device__ __forceinline__ v8f wmma_bf16(v16bf a, v16bf b, v8f c) {
    // (neg_a, A, neg_b, B, c_mod, C, reuse_a, reuse_b)
    return __builtin_amdgcn_wmma_f32_16x16x32_bf16(false, a, false, b,
                                                   (short)0, c, false, false);
}

// ---------------------------------------------------------------------------
// Problem constants
// ---------------------------------------------------------------------------
#define CN    4096      // tokens (S*L)
#define CHID  4096
#define CH    32        // q heads
#define CKV   8         // kv heads
#define CD    128       // head dim
#define CHD   4096      // H*D
#define CKVD  1024      // KV*D
#define CL    1024      // tokens per seq

// ---------------------------------------------------------------------------
// GEMM: C[N x Mout] = X[N x K] * W[Mout x K]^T  (+ X2 * W2^T, fused 2nd pass)
// 64x64 block tile, 8 waves (wave32), each wave -> two 16x16 f32 accumulators.
// Register + LDS double buffering: global loads of tile t+1 are issued while
// the WMMAs of tile t run; one barrier per iteration (ping-pong safe).
// ---------------------------------------------------------------------------
__global__ __launch_bounds__(256) void wmma_gemm_xwT(
    const float* __restrict__ X,  const float* __restrict__ W,
    const float* __restrict__ X2, const float* __restrict__ W2,
    float* __restrict__ C, int K, int Mout)
{
    __shared__ unsigned short lA[2][64 * 32];
    __shared__ unsigned short lB[2][64 * 32];

    const int tid  = threadIdx.x;
    const int lane = tid & 31;
    const int wv   = tid >> 5;       // 0..7
    const int mt   = wv & 3;         // row tile 0..3
    const int np   = wv >> 2;        // col pair 0..1
    const int bm   = blockIdx.y * 64;
    const int bn   = blockIdx.x * 64;

    const int mrow  = lane & 15;
    const int koffA = (lane >> 4) << 3;   // A 16-bit layout: 0 / 8
    const int koffB = (lane >> 4) << 4;   // unused: B fragments are read with koffA (0 / 8)

    // staging geometry: each thread owns 2 float4 per matrix per tile
    const int row0 = tid >> 3;            // 0..31
    const int row1 = row0 + 32;           // 32..63
    const int c4   = (tid & 7) << 2;      // 0..28

    const int ktiles = K >> 5;
    const int total  = (X2 ? 2 : 1) * ktiles;

    v8f acc0 = {};
    v8f acc1 = {};

    float4 ra0, ra1, rb0, rb1;
    // prologue: tile 0
    ra0 = *(const float4*)(X + (size_t)(bm + row0) * K + c4);
    ra1 = *(const float4*)(X + (size_t)(bm + row1) * K + c4);
    rb0 = *(const float4*)(W + (size_t)(bn + row0) * K + c4);
    rb1 = *(const float4*)(W + (size_t)(bn + row1) * K + c4);

    int buf = 0;
    for (int t = 0; t < total; ++t) {
        // --- convert staged registers to bf16, store to LDS[buf] ----------
        {
            uint2 ua0 = {f2bf2(ra0.x, ra0.y), f2bf2(ra0.z, ra0.w)};
            uint2 ua1 = {f2bf2(ra1.x, ra1.y), f2bf2(ra1.z, ra1.w)};
            uint2 ub0 = {f2bf2(rb0.x, rb0.y), f2bf2(rb0.z, rb0.w)};
            uint2 ub1 = {f2bf2(rb1.x, rb1.y), f2bf2(rb1.z, rb1.w)};
            *(uint2*)&lA[buf][row0 * 32 + c4] = ua0;
            *(uint2*)&lA[buf][row1 * 32 + c4] = ua1;
            *(uint2*)&lB[buf][row0 * 32 + c4] = ub0;
            *(uint2*)&lB[buf][row1 * 32 + c4] = ub1;
        }
        __syncthreads();

        // --- issue next tile's global loads (overlap with WMMAs below) ----
        if (t + 1 < total) {
            const int  tn = t + 1;
            const bool p2 = tn >= ktiles;
            const float* Xp = p2 ? X2 : X;
            const float* Wp = p2 ? W2 : W;
            const int  k0 = (p2 ? tn - ktiles : tn) << 5;
            ra0 = *(const float4*)(Xp + (size_t)(bm + row0) * K + k0 + c4);
            ra1 = *(const float4*)(Xp + (size_t)(bm + row1) * K + k0 + c4);
            rb0 = *(const float4*)(Wp + (size_t)(bn + row0) * K + k0 + c4);
            rb1 = *(const float4*)(Wp + (size_t)(bn + row1) * K + k0 + c4);
        }

        // --- fragments per documented CDNA5 layouts + WMMA ----------------
        Frag16 a;
        a.u8[0] = *(const ushort8*)&lA[buf][(mt * 16 + mrow) * 32 + koffA];
        a.u8[1] = *(const ushort8*)&lA[buf][(mt * 16 + mrow) * 32 + 16 + koffA];

        Frag16 b0, b1;
        b0.u8[0] = *(const ushort8*)&lB[buf][((np * 2 + 0) * 16 + mrow) * 32 + koffA];
        b0.u8[1] = *(const ushort8*)&lB[buf][((np * 2 + 0) * 16 + mrow) * 32 + 16 + koffA];
        b1.u8[0] = *(const ushort8*)&lB[buf][((np * 2 + 1) * 16 + mrow) * 32 + koffA];
        b1.u8[1] = *(const ushort8*)&lB[buf][((np * 2 + 1) * 16 + mrow) * 32 + 16 + koffA];

        acc0 = wmma_bf16(a.v, b0.v, acc0);
        acc1 = wmma_bf16(a.v, b1.v, acc1);

        buf ^= 1;
    }

    // --- epilogue: C/D 16x16 f32 layout -----------------------------------
    const int rbase = bm + mt * 16 + ((lane >> 4) << 3);
    const int c0    = bn + (np * 2 + 0) * 16 + mrow;
    const int c1    = bn + (np * 2 + 1) * 16 + mrow;
    #pragma unroll
    for (int r = 0; r < 8; ++r) {
        C[(size_t)(rbase + r) * Mout + c0] = acc0[r];
        C[(size_t)(rbase + r) * Mout + c1] = acc1[r];
    }
}

// ---------------------------------------------------------------------------
// Fused per-head RMSNorm + RoPE, in place on f32 [N, heads, 128].
// One wave per (token, head); lane holds elems {l, l+32, l+64, l+96} so each
// rotate_half pair (i, i+64) stays in-lane.
// ---------------------------------------------------------------------------
__global__ __launch_bounds__(256) void rmsnorm_rope(
    float* __restrict__ X, const float* __restrict__ w,
    const int* __restrict__ pos, int heads)
{
    const int lane = threadIdx.x & 31;
    const int wv   = threadIdx.x >> 5;
    const int pair = blockIdx.x * 8 + wv;        // tok*heads + head
    const int tok  = pair / heads;

    float* xp = X + (size_t)pair * CD;
    float x0 = xp[lane];
    float x1 = xp[lane + 32];
    float x2 = xp[lane + 64];
    float x3 = xp[lane + 96];

    float ss = x0 * x0 + x1 * x1 + x2 * x2 + x3 * x3;
    ss += __shfl_xor(ss, 1, 32);
    ss += __shfl_xor(ss, 2, 32);
    ss += __shfl_xor(ss, 4, 32);
    ss += __shfl_xor(ss, 8, 32);
    ss += __shfl_xor(ss, 16, 32);
    const float rn = rsqrtf(ss * (1.0f / 128.0f) + 1e-6f);

    x0 *= rn * w[lane];
    x1 *= rn * w[lane + 32];
    x2 *= rn * w[lane + 64];
    x3 *= rn * w[lane + 96];

    const float p        = (float)pos[tok];
    const float ln_theta = 9.210340371976184f;   // ln(10000)
    const float a0 = p * __expf(-(float)lane * (ln_theta / 64.0f));
    const float a1 = p * __expf(-((float)lane + 32.0f) * (ln_theta / 64.0f));
    const float c0 = __cosf(a0), s0 = __sinf(a0);
    const float c1 = __cosf(a1), s1 = __sinf(a1);

    xp[lane]      = x0 * c0 - x2 * s0;
    xp[lane + 64] = x2 * c0 + x0 * s0;
    xp[lane + 32] = x1 * c1 - x3 * s1;
    xp[lane + 96] = x3 * c1 + x1 * s1;
}

// ---------------------------------------------------------------------------
// Flash attention (causal, GQA 4:1). Block = 4 waves = 64 queries of one
// (seq, head). 32-key blocks staged to LDS as bf16; scores & PV via WMMA;
// online softmax with shfl reductions matching the C-matrix row layout.
// ---------------------------------------------------------------------------
__global__ __launch_bounds__(128) void flash_attn(
    const float* __restrict__ Q, const float* __restrict__ Kc,
    const float* __restrict__ V, float* __restrict__ O)
{
    __shared__ unsigned short lK[32 * 128];      // [key][d]  bf16
    __shared__ unsigned short lVt[128 * 32];     // [d][key]  bf16 (transposed)
    __shared__ unsigned short lP[4][16 * 32];    // per-wave P tile

    const int tid  = threadIdx.x;
    const int lane = tid & 31;
    const int wv   = tid >> 5;
    const int qb   = blockIdx.x * 64;
    const int h    = blockIdx.y;
    const int s    = blockIdx.z;
    const int kvh  = h >> 2;                     // GROUPS = 4
    const int q0   = qb + wv * 16;

    const int mrow  = lane & 15;
    const int koffA = (lane >> 4) << 3;          // 0 / 8
    const int koffB = (lane >> 4) << 4;          // 0 / 16
    const float scale = 0.08838834764831845f;    // 1/sqrt(128)

    // --- preload Q fragments (scale folded in), packed conversion ---------
    const float* qp = Q + (size_t)(s * CL + q0 + mrow) * CHD + h * CD;
    Frag16 fq[4];
    #pragma unroll
    for (int c = 0; c < 4; ++c) {
        const float4 qa = *(const float4*)(qp + c * 32 + koffA);
        const float4 qb2 = *(const float4*)(qp + c * 32 + koffA + 4);
        const float4 qc = *(const float4*)(qp + c * 32 + koffA + 16);
        const float4 qd = *(const float4*)(qp + c * 32 + koffA + 20);
        fq[c].u32[0] = f2bf2(qa.x * scale, qa.y * scale);
        fq[c].u32[1] = f2bf2(qa.z * scale, qa.w * scale);
        fq[c].u32[2] = f2bf2(qb2.x * scale, qb2.y * scale);
        fq[c].u32[3] = f2bf2(qb2.z * scale, qb2.w * scale);
        fq[c].u32[4] = f2bf2(qc.x * scale, qc.y * scale);
        fq[c].u32[5] = f2bf2(qc.z * scale, qc.w * scale);
        fq[c].u32[6] = f2bf2(qd.x * scale, qd.y * scale);
        fq[c].u32[7] = f2bf2(qd.z * scale, qd.w * scale);
    }

    float mmax[8], lsum[8];
    v8f o[8];
    const v8f vzero = {};
    #pragma unroll
    for (int r = 0; r < 8; ++r) { mmax[r] = -1e30f; lsum[r] = 0.f; }
    #pragma unroll
    for (int t = 0; t < 8; ++t) o[t] = vzero;

    const int kend = qb + 64;
    for (int kb = 0; kb < kend; kb += 32) {
        // --- stage K/V block (32 keys x 128 dims) to LDS as bf16 ----------
        #pragma unroll
        for (int i = 0; i < 8; ++i) {
            int idx = tid + i * 128;             // 0..1023
            int key = idx >> 5;                  // 0..31
            int c4  = (idx & 31) << 2;           // 0..124
            size_t gro = (size_t)(s * CL + kb + key) * CKVD + kvh * CD + c4;
            const float4 fk = *(const float4*)(Kc + gro);
            const float4 fv = *(const float4*)(V + gro);
            uint2 pk = {f2bf2(fk.x, fk.y), f2bf2(fk.z, fk.w)};
            *(uint2*)&lK[key * 128 + c4] = pk;
            lVt[(c4 + 0) * 32 + key] = f2bf(fv.x);
            lVt[(c4 + 1) * 32 + key] = f2bf(fv.y);
            lVt[(c4 + 2) * 32 + key] = f2bf(fv.z);
            lVt[(c4 + 3) * 32 + key] = f2bf(fv.w);
            if (kb + 32 < kend) {                // prefetch next key block
                __builtin_prefetch(Kc + gro + 32 * CKVD, 0, 3);
                __builtin_prefetch(V  + gro + 32 * CKVD, 0, 3);
            }
        }
        __syncthreads();

        // --- scores: S[16x32] = Q[16x128] * K^T, 2 col-tiles x 4 K-chains -
        v8f sc[2];
        sc[0] = vzero; sc[1] = vzero;
        #pragma unroll
        for (int t = 0; t < 2; ++t) {
            #pragma unroll
            for (int c = 0; c < 4; ++c) {
                Frag16 b;
                b.u8[0] = *(const ushort8*)&lK[(t * 16 + mrow) * 128 + c * 32 + koffA];
                b.u8[1] = *(const ushort8*)&lK[(t * 16 + mrow) * 128 + c * 32 + 16 + koffA];
                sc[t] = wmma_bf16(fq[c].v, b.v, sc[t]);
            }
        }

        // --- causal mask + online softmax (rows live in 16-lane halves) ---
        const int qrow0 = q0 + ((lane >> 4) << 3);
        const int phr   = (lane >> 4) << 3;
        #pragma unroll
        for (int r = 0; r < 8; ++r) {
            const int qpos = qrow0 + r;
            float s0 = sc[0][r]; if (kb + mrow      > qpos) s0 = -1e30f;
            float s1 = sc[1][r]; if (kb + 16 + mrow > qpos) s1 = -1e30f;

            float mx = fmaxf(s0, s1);
            mx = fmaxf(mx, __shfl_xor(mx, 1, 32));
            mx = fmaxf(mx, __shfl_xor(mx, 2, 32));
            mx = fmaxf(mx, __shfl_xor(mx, 4, 32));
            mx = fmaxf(mx, __shfl_xor(mx, 8, 32));

            const float mnew  = fmaxf(mmax[r], mx);
            const float alpha = __expf(mmax[r] - mnew);
            mmax[r] = mnew;

            const float p0 = __expf(s0 - mnew);
            const float p1 = __expf(s1 - mnew);
            float ps = p0 + p1;
            ps += __shfl_xor(ps, 1, 32);
            ps += __shfl_xor(ps, 2, 32);
            ps += __shfl_xor(ps, 4, 32);
            ps += __shfl_xor(ps, 8, 32);
            lsum[r] = lsum[r] * alpha + ps;

            #pragma unroll
            for (int t = 0; t < 8; ++t) o[t][r] *= alpha;

            lP[wv][(phr + r) * 32 + mrow]      = f2bf(p0);
            lP[wv][(phr + r) * 32 + 16 + mrow] = f2bf(p1);
        }
        __builtin_amdgcn_wave_barrier();   // keep P stores before P reads

        // --- P fragment (A layout) and PV accumulate ----------------------
        Frag16 fp;
        fp.u8[0] = *(const ushort8*)&lP[wv][mrow * 32 + koffA];
        fp.u8[1] = *(const ushort8*)&lP[wv][mrow * 32 + 16 + koffA];

        #pragma unroll
        for (int t = 0; t < 8; ++t) {
            Frag16 b;
            b.u8[0] = *(const ushort8*)&lVt[(t * 16 + mrow) * 32 + koffA];
            b.u8[1] = *(const ushort8*)&lVt[(t * 16 + mrow) * 32 + 16 + koffA];
            o[t] = wmma_bf16(fp.v, b.v, o[t]);
        }
        __syncthreads();
    }

    // --- epilogue: normalize and store [N, H*D] f32 -----------------------
    const int rowoff = (lane >> 4) << 3;
    #pragma unroll
    for (int r = 0; r < 8; ++r) {
        const float inv = 1.0f / lsum[r];
        const size_t base = (size_t)(s * CL + q0 + rowoff + r) * CHD + h * CD;
        #pragma unroll
        for (int t = 0; t < 8; ++t) {
            O[base + t * 16 + mrow] = o[t][r] * inv;
        }
    }
}

// ---------------------------------------------------------------------------
// Host launcher
// ---------------------------------------------------------------------------
extern "C" void kernel_launch(void* const* d_in, const int* in_sizes, int n_in,
                              void* d_out, int out_size, void* d_ws, size_t ws_size,
                              hipStream_t stream)
{
    const float* hidden = (const float*)d_in[0];
    const float* mu     = (const float*)d_in[1];
    const int*   pos    = (const int*)d_in[2];
    const float* Wq     = (const float*)d_in[3];
    const float* Wk     = (const float*)d_in[4];
    const float* Wv     = (const float*)d_in[5];
    const float* Wo     = (const float*)d_in[6];
    const float* Wmq    = (const float*)d_in[7];
    const float* Wmk    = (const float*)d_in[8];
    const float* Wmv    = (const float*)d_in[9];
    const float* qnw    = (const float*)d_in[10];
    const float* knw    = (const float*)d_in[11];
    float* out = (float*)d_out;

    char* ws = (char*)d_ws;
    float* qf = (float*)(ws);                                                    // 64 MB
    float* kf = (float*)(ws + (size_t)CN * CHD * 4);                             // 16 MB
    float* vf = (float*)(ws + (size_t)CN * CHD * 4 + (size_t)CN * CKVD * 4);     // 16 MB
    float* of = (float*)(ws + (size_t)CN * CHD * 4 + 2 * (size_t)CN * CKVD * 4); // 64 MB

    const dim3 blk(256);
    // QKV projections with mu-bias fusion (second K-pass)
    wmma_gemm_xwT<<<dim3(CHD / 64, CN / 64), blk, 0, stream>>>(hidden, Wq, mu, Wmq, qf, CHID, CHD);
    wmma_gemm_xwT<<<dim3(CKVD / 64, CN / 64), blk, 0, stream>>>(hidden, Wk, mu, Wmk, kf, CHID, CKVD);
    wmma_gemm_xwT<<<dim3(CKVD / 64, CN / 64), blk, 0, stream>>>(hidden, Wv, mu, Wmv, vf, CHID, CKVD);
    // QK RMSNorm + RoPE (in place)
    rmsnorm_rope<<<CN * CH  / 8, blk, 0, stream>>>(qf, qnw, pos, CH);
    rmsnorm_rope<<<CN * CKV / 8, blk, 0, stream>>>(kf, knw, pos, CKV);
    // causal GQA flash attention
    flash_attn<<<dim3(CL / 64, CH, CN / CL), dim3(128), 0, stream>>>(qf, kf, vf, of);
    // output projection straight into d_out
    wmma_gemm_xwT<<<dim3(CHID / 64, CN / 64), blk, 0, stream>>>(of, Wo, nullptr, nullptr, out, CHD, CHID);
}
